// DecoderLayer_28217935135362
// MI455X (gfx1250) — matrix operand order repaired and checked by reading, not compiled, not run on hardware
//
#include <hip/hip_runtime.h>
#include <hip/hip_bf16.h>

// ---------------------------------------------------------------------------
// Problem constants (from the reference): B=16, S=512, D=512, H=8, DH=64
// ---------------------------------------------------------------------------
#define BB   16
#define SS   512
#define DD   512
#define HH   8
#define DHD  64
#define MROWS (BB * SS)          // 8192 global rows
#define NEGBIG (-1e9f)

typedef __attribute__((ext_vector_type(16))) _Float16 v16h;
typedef __attribute__((ext_vector_type(8)))  _Float16 v8h;
typedef __attribute__((ext_vector_type(8)))  float    v8f;

// ---------------------------------------------------------------------------
// WMMA helpers (CDNA5: V_WMMA_F32_16X16X32_F16, wave32)
// ---------------------------------------------------------------------------
__device__ __forceinline__ v8f wmma32(v16h a, v16h b, v8f c) {
  // 8 args: (neg_a, A, neg_b, B, c_mod, C, reuse_a, reuse_b)
  return __builtin_amdgcn_wmma_f32_16x16x32_f16(false, a, false, b,
                                                (short)0, c, false, false);
}

// Wave-local LDS ordering: DS ops from one wave are in-order (ISA Ch.11) and
// tracked by DScnt; s_wait_dscnt 0 + compiler scheduling barriers make the
// per-wave LDS transpose safe without a block-wide s_barrier.
__device__ __forceinline__ void wave_lds_fence() {
  __builtin_amdgcn_wave_barrier();
  asm volatile("s_wait_dscnt 0" ::: "memory");
  __builtin_amdgcn_wave_barrier();
}

// A-matrix fragment (16x32 f16, MxK), row-major source with leading dim `ld`.
// ISA layout: lane m = lane&15; hi half selects K 8-15 / 24-31.
__device__ __forceinline__ v16h load_a_frag(const _Float16* __restrict__ base,
                                            int ld, int m0, int k0, int lane) {
  const int m  = lane & 15;
  const int hi = lane >> 4;
  const _Float16* row = base + (size_t)(m0 + m) * ld + k0;
  v16h a;
#pragma unroll
  for (int j = 0; j < 8; ++j) {
    const int kb = ((j & 4) << 2) + hi * 8 + ((j & 3) << 1); // (j<4?0:16)+hi*8+2*(j&3)
    a[2 * j]     = row[kb];
    a[2 * j + 1] = row[kb + 1];
  }
  return a;
}

// B-matrix fragment (32x16 f16, KxN): lane n = lane&15; hi half selects K 8-15 / 24-31.
// Strided gather per lane from row-major [K x N] with leading dim ld.
__device__ __forceinline__ v16h load_b_frag(const _Float16* __restrict__ base,
                                            int ld, int k0, int n0, int lane) {
  const int n  = lane & 15;
  const int hi = lane >> 4;
  const _Float16* col = base + (size_t)k0 * ld + n0 + n;
  v16h b;
#pragma unroll
  for (int j = 0; j < 8; ++j) {
    const int kb = ((j & 4) << 2) + hi * 8 + ((j & 3) << 1);
    b[2 * j]     = col[(size_t)kb * ld];
    b[2 * j + 1] = col[(size_t)(kb + 1) * ld];
  }
  return b;
}

// ---------------------------------------------------------------------------
// Conversion kernels (pure bandwidth; 8 elems/thread, b128 traffic)
// ---------------------------------------------------------------------------
__global__ void cvt_f32_to_f16_kernel(const float* __restrict__ in,
                                      _Float16* __restrict__ out, int n8) {
  int i = blockIdx.x * blockDim.x + threadIdx.x;
  if (i >= n8) return;
  const v8f x = *reinterpret_cast<const v8f*>(in + 8 * (size_t)i);
  v8h y;
#pragma unroll
  for (int j = 0; j < 8; ++j) y[j] = (_Float16)x[j];
  *reinterpret_cast<v8h*>(out + 8 * (size_t)i) = y;
}

// Wq/Wk/Wv: [H, D, DH] f32 -> [D, H*DH] f16 (column n = h*64+e), scaled.
// Contiguous over e, so vectorize by 8.
__global__ void cvt_w_head_kernel(const float* __restrict__ in,
                                  _Float16* __restrict__ out, float scale) {
  int t = blockIdx.x * blockDim.x + threadIdx.x;   // t in [0, 512*512/8)
  const int d  = t >> 6;
  const int r  = t & 63;
  const int h  = r >> 3;
  const int e0 = (r & 7) * 8;
  const v8f x = *reinterpret_cast<const v8f*>(in + ((size_t)h * DD + d) * DHD + e0);
  v8h y;
#pragma unroll
  for (int j = 0; j < 8; ++j) y[j] = (_Float16)(x[j] * scale);
  *reinterpret_cast<v8h*>(out + (size_t)d * DD + h * DHD + e0) = y;
}

// ---------------------------------------------------------------------------
// Projection GEMM: Y[M=8192, N=512] = X[8192,512](f16) @ W[512,512](f16)
// One wave computes a 16x64 tile; 8 waves/block share the same 16 A-rows
// (consecutive tiles cover N) for L1/L2 reuse. W (512 KB) is L2-resident.
// MODE 0: write f16 [B,H,S,DH]   (Q / V layout)
// MODE 1: write f16 [B,H,DH,S]   (K transposed for score B-fragments)
// MODE 2: write f32 [B,S,D] + bias (final output)
// ---------------------------------------------------------------------------
template <int MODE>
__global__ void gemm_proj_kernel(const _Float16* __restrict__ X,
                                 const _Float16* __restrict__ W,
                                 void* __restrict__ Y,
                                 const float* __restrict__ bias) {
  const int lane = threadIdx.x & 31;
  const int wave = threadIdx.x >> 5;
  const int tile = blockIdx.x * 8 + wave;          // 4096 wave-tiles total
  const int colTile = tile & 7;                    // N/64 = 8
  const int rowTile = tile >> 3;                   // M/16 = 512
  const int m0 = rowTile * 16;
  const int n0 = colTile * 64;

  v8f c0 = {}, c1 = {}, c2 = {}, c3 = {};
#pragma unroll 4
  for (int k0 = 0; k0 < DD; k0 += 32) {
    // prefetch next A chunk (lowers to global_prefetch_b8 on gfx1250)
    if (k0 + 64 < DD)
      __builtin_prefetch(X + (size_t)(m0 + (lane & 15)) * DD + k0 + 64, 0, 1);
    v16h a = load_a_frag(X, DD, m0, k0, lane);
    v16h b0 = load_b_frag(W, DD, k0, n0 +  0, lane);
    v16h b1 = load_b_frag(W, DD, k0, n0 + 16, lane);
    v16h b2 = load_b_frag(W, DD, k0, n0 + 32, lane);
    v16h b3 = load_b_frag(W, DD, k0, n0 + 48, lane);
    c0 = wmma32(a, b0, c0);
    c1 = wmma32(a, b1, c1);
    c2 = wmma32(a, b2, c2);
    c3 = wmma32(a, b3, c3);
  }

  const int hi = lane >> 4;
  const int nn = lane & 15;
  v8f acc[4] = {c0, c1, c2, c3};
#pragma unroll
  for (int f = 0; f < 4; ++f) {
    const int col = n0 + f * 16 + nn;              // n = h*64 + e
#pragma unroll
    for (int r = 0; r < 8; ++r) {
      const int g = m0 + r + 8 * hi;               // global row = b*S + s
      const float val = acc[f][r];
      if (MODE == 2) {
        ((float*)Y)[(size_t)g * DD + col] = val + bias[col];
      } else {
        const int bb = g >> 9, s = g & 511;
        const int h = col >> 6, e = col & 63;
        if (MODE == 0)
          ((_Float16*)Y)[(((size_t)bb * HH + h) * SS + s) * DHD + e] = (_Float16)val;
        else
          ((_Float16*)Y)[(((size_t)bb * HH + h) * DHD + e) * SS + s] = (_Float16)val;
      }
    }
  }
}

// ---------------------------------------------------------------------------
// Flash attention: per (b,h, 64-row block). 4 independent waves x 16 rows.
// Q: [B,H,S,DH] f16 (score scale folded into Wq), Kt: [B,H,DH,S] f16,
// V: [B,H,S,DH] f16.  Output: f16 [B,S,D], column h*64+e (head concat).
// Online softmax; P never touches HBM (O(S*DH) traffic/head, not O(S^2)).
// ---------------------------------------------------------------------------
__global__ void flash_attn_kernel(const _Float16* __restrict__ Q,
                                  const _Float16* __restrict__ Kt,
                                  const _Float16* __restrict__ V,
                                  _Float16* __restrict__ Hout,
                                  int causal) {
  __shared__ _Float16 Plds[4][16][32];             // per-wave P staging (4 KB)

  const int lane = threadIdx.x & 31;
  const int wave = threadIdx.x >> 5;
  const int bh   = blockIdx.x >> 3;                // b*H + h
  const int rblk = blockIdx.x & 7;                 // S/64 row blocks
  const int bb   = bh >> 3;
  const int h    = bh & 7;
  const int srow0 = rblk * 64 + wave * 16;
  const int hi = lane >> 4;
  const int nn = lane & 15;

  const _Float16* Qbh  = Q  + (size_t)bh * SS * DHD;
  const _Float16* Ktbh = Kt + (size_t)bh * DHD * SS;
  const _Float16* Vbh  = V  + (size_t)bh * SS * DHD;

  // Q fragments for the wave's 16 rows: DH=64 -> two K=32 chunks.
  const v16h aq0 = load_a_frag(Qbh, DHD, srow0, 0,  lane);
  const v16h aq1 = load_a_frag(Qbh, DHD, srow0, 32, lane);

  v8f acc[4] = {};                                 // 16x64 f32 output accum
  float mrow[8], lrow[8];
#pragma unroll
  for (int r = 0; r < 8; ++r) { mrow[r] = -3e38f; lrow[r] = 0.0f; }

  // Waves are independent (no block barrier), so each stops at its own
  // causal frontier: rows srow0..srow0+15 need keys t <= srow0+15.
  const int t_limit = causal ? (srow0 + 16) : SS;

  for (int t0 = 0; t0 < t_limit; t0 += 32) {
    // ---- scores: two 16x16 tiles (keys t0..t0+15 and t0+16..t0+31) ----
    v8f sc0 = {}, sc1 = {};
    {
      v16h bk = load_b_frag(Ktbh, SS, 0, t0, lane);        // e-chunk 0
      sc0 = wmma32(aq0, bk, sc0);
      bk = load_b_frag(Ktbh, SS, 0, t0 + 16, lane);
      sc1 = wmma32(aq0, bk, sc1);
      bk = load_b_frag(Ktbh, SS, 32, t0, lane);            // e-chunk 1
      sc0 = wmma32(aq1, bk, sc0);
      bk = load_b_frag(Ktbh, SS, 32, t0 + 16, lane);
      sc1 = wmma32(aq1, bk, sc1);
    }

    // ---- online softmax update per owned row (r -> row m = r + 8*hi) ----
#pragma unroll
    for (int r = 0; r < 8; ++r) {
      const int s = srow0 + r + 8 * hi;
      float v0 = sc0[r], v1 = sc1[r];
      if (causal) {
        const int t = t0 + nn;
        if (t > s)      v0 = NEGBIG;
        if (t + 16 > s) v1 = NEGBIG;
      }
      float x = fmaxf(v0, v1);
#pragma unroll
      for (int off = 8; off > 0; off >>= 1)        // max over 16-lane group
        x = fmaxf(x, __shfl_xor(x, off, 32));
      const float mprev = mrow[r];
      const float mnew  = fmaxf(mprev, x);
      const float alpha = __expf(mprev - mnew);
      const float p0 = __expf(v0 - mnew);
      const float p1 = __expf(v1 - mnew);
      float sum = p0 + p1;
#pragma unroll
      for (int off = 8; off > 0; off >>= 1)
        sum += __shfl_xor(sum, off, 32);
      lrow[r] = lrow[r] * alpha + sum;
      mrow[r] = mnew;
#pragma unroll
      for (int f = 0; f < 4; ++f) acc[f][r] *= alpha;
      Plds[wave][r + 8 * hi][nn]      = (_Float16)p0;
      Plds[wave][r + 8 * hi][16 + nn] = (_Float16)p1;
    }
    wave_lds_fence();                              // P stores -> A-frag loads

    // ---- P (16x32, A layout) @ V tile (32x64) ----
    const v16h ap = load_a_frag(&Plds[wave][0][0], 32, 0, 0, lane);
    wave_lds_fence();                              // loads before next store
#pragma unroll
    for (int f = 0; f < 4; ++f) {
      const v16h bv = load_b_frag(Vbh, DHD, t0, f * 16, lane);
      acc[f] = wmma32(ap, bv, acc[f]);
    }
  }

  // ---- normalize and write head-concat output [B,S,D] ----
#pragma unroll
  for (int r = 0; r < 8; ++r) {
    const int s = srow0 + r + 8 * hi;
    const float inv = 1.0f / lrow[r];
#pragma unroll
    for (int f = 0; f < 4; ++f) {
      const int e = f * 16 + nn;
      Hout[((size_t)bb * SS + s) * DD + h * DHD + e] = (_Float16)(acc[f][r] * inv);
    }
  }
}

// ---------------------------------------------------------------------------
// Host-side orchestration
// ---------------------------------------------------------------------------
extern "C" void kernel_launch(void* const* d_in, const int* in_sizes, int n_in,
                              void* d_out, int out_size, void* d_ws, size_t ws_size,
                              hipStream_t stream) {
  (void)in_sizes; (void)n_in; (void)out_size;
  const float* de_x = (const float*)d_in[0];
  const float* en_x = (const float*)d_in[1];
  // d_in[2] = mask (triu, k=1): implemented analytically as causal masking
  const float* Wq = (const float*)d_in[3];
  const float* Wk = (const float*)d_in[4];
  const float* Wv = (const float*)d_in[5];
  const float* W2 = (const float*)d_in[6];
  const float* b2 = (const float*)d_in[7];
  float* out = (float*)d_out;

  // Workspace layout (f16 elements). Needs 50 MB.
  const size_t NX = (size_t)MROWS * DD;            // 4,194,304
  const size_t NW = (size_t)DD * DD;               // 262,144
  const size_t need_bytes = (2 * NX + 4 * NW + 4 * NX) * sizeof(_Float16);
  if (ws_size < need_bytes) return;                // insufficient scratch

  _Float16* ws   = (_Float16*)d_ws;
  _Float16* de16 = ws;
  _Float16* en16 = de16 + NX;
  _Float16* Wq16 = en16 + NX;
  _Float16* Wk16 = Wq16 + NW;
  _Float16* Wv16 = Wk16 + NW;
  _Float16* W216 = Wv16 + NW;
  _Float16* Qb   = W216 + NW;                      // reused by block 1 & 2
  _Float16* Ktb  = Qb + NX;
  _Float16* Vb   = Ktb + NX;
  _Float16* h1   = Vb + NX;
  _Float16* h2   = de16;                           // de16 dead after projections

  const int CT = 256;
  // --- convert inputs / weights to f16 (8 elems/thread) ---
  cvt_f32_to_f16_kernel<<<(int)(NX / 8 / CT), CT, 0, stream>>>(de_x, de16, (int)(NX / 8));
  cvt_f32_to_f16_kernel<<<(int)(NX / 8 / CT), CT, 0, stream>>>(en_x, en16, (int)(NX / 8));
  cvt_f32_to_f16_kernel<<<(int)(NW / 8 / CT), CT, 0, stream>>>(W2, W216, (int)(NW / 8));
  const float qscale = 0.125f;                     // 1/sqrt(DH) folded into Wq
  cvt_w_head_kernel<<<(int)(NW / 8 / CT), CT, 0, stream>>>(Wq, Wq16, qscale);
  cvt_w_head_kernel<<<(int)(NW / 8 / CT), CT, 0, stream>>>(Wk, Wk16, 1.0f);
  cvt_w_head_kernel<<<(int)(NW / 8 / CT), CT, 0, stream>>>(Wv, Wv16, 1.0f);

  const int GEMM_BLOCKS = (MROWS / 16) * (DD / 64) / 8;  // 512
  const int ATTN_BLOCKS = BB * HH * (SS / 64);           // 1024

  // --- block 1: projections from de_x, causal attention ---
  gemm_proj_kernel<0><<<GEMM_BLOCKS, 256, 0, stream>>>(de16, Wq16, Qb, nullptr);
  gemm_proj_kernel<1><<<GEMM_BLOCKS, 256, 0, stream>>>(de16, Wk16, Ktb, nullptr);
  gemm_proj_kernel<0><<<GEMM_BLOCKS, 256, 0, stream>>>(de16, Wv16, Vb, nullptr);
  flash_attn_kernel<<<ATTN_BLOCKS, 128, 0, stream>>>(Qb, Ktb, Vb, h1, 1);

  // --- block 2: q,k from en_x; v from h1; no mask ---
  gemm_proj_kernel<0><<<GEMM_BLOCKS, 256, 0, stream>>>(en16, Wq16, Qb, nullptr);
  gemm_proj_kernel<1><<<GEMM_BLOCKS, 256, 0, stream>>>(en16, Wk16, Ktb, nullptr);
  gemm_proj_kernel<0><<<GEMM_BLOCKS, 256, 0, stream>>>(h1, Wv16, Vb, nullptr);
  flash_attn_kernel<<<ATTN_BLOCKS, 128, 0, stream>>>(Qb, Ktb, Vb, h2, 0);

  // --- final linear: out = h2 @ W2 + b2 (f32) ---
  gemm_proj_kernel<2><<<GEMM_BLOCKS, 256, 0, stream>>>(h2, W216, out, b2);
}
